// SDNN_18279380812278
// MI455X (gfx1250) — matrix operand order repaired and checked by reading, not compiled, not run on hardware
//
#include <hip/hip_runtime.h>

// ===========================================================================
// SDNN forward for MI455X (gfx1250, wave32, WMMA 16x16x32 f16).
// Split-f16 ("3xFP16") GEMMs; all operands pre-converted ONCE into hi/lo f16
// planes laid out in WMMA fragment order, so the hot loops are pure
// global_load_b128 + v_wmma.  No LDS staging in the GEMM: the 192MB L2 keeps
// the packed working set (~150MB) resident.  Wave tile 64x64 -> 48 WMMAs per
// 32 b128 loads per K-tile.
// ===========================================================================

typedef __attribute__((ext_vector_type(16))) _Float16 v16h;
typedef __attribute__((ext_vector_type(8)))  float    v8f;
typedef unsigned int u32;

union H2   { _Float16 f[2]; u32 u; };       // two f16 packed in a dword
union Frag { uint4 u4[2]; v16h h; };        // 8 dwords = one 16x32-f16 fragment

// 3-term split product: hi*hi + hi*lo + lo*hi  (~fp32 accuracy)
__device__ __forceinline__ v8f wmma3(v16h ah, v16h al, v16h bh, v16h bl, v8f c) {
  c = __builtin_amdgcn_wmma_f32_16x16x32_f16(false, ah, false, bh, (short)0, c, false, false);
  c = __builtin_amdgcn_wmma_f32_16x16x32_f16(false, ah, false, bl, (short)0, c, false, false);
  c = __builtin_amdgcn_wmma_f32_16x16x32_f16(false, al, false, bh, (short)0, c, false, false);
  return c;
}

__device__ __forceinline__ void split1(float f, _Float16& hi, _Float16& lo) {
  hi = (_Float16)f;
  lo = (_Float16)(f - (float)hi);
}

// ===========================================================================
// Pack kernels (run once; bandwidth-trivial vs 23.3 TB/s)
// ===========================================================================

// A operand: fp32 [M][K] -> plain row-major f16 hi/lo planes.
// A-fragment K pairs {2v,2v+1} are contiguous, so a fragment is 2x b128/plane.
__global__ __launch_bounds__(256) void pack_a(const float* __restrict__ A,
                                              u32* __restrict__ Ah, u32* __restrict__ Al,
                                              size_t n4 /* total/4 */) {
  size_t g = (size_t)blockIdx.x * 256 + threadIdx.x;
  if (g >= n4) return;
  float4 f = reinterpret_cast<const float4*>(A)[g];
  H2 h0, h1, l0, l1;
  split1(f.x, h0.f[0], l0.f[0]); split1(f.y, h0.f[1], l0.f[1]);
  split1(f.z, h1.f[0], l1.f[0]); split1(f.w, h1.f[1], l1.f[1]);
  reinterpret_cast<uint2*>(Ah)[g] = (uint2){h0.u, h1.u};
  reinterpret_cast<uint2*>(Al)[g] = (uint2){l0.u, l1.u};
}

// B operand: fp32 [K][ldb] (x mask) -> hi/lo planes in B-fragment order:
// dword index = ((kt*N + n)*2 + half)*8 + v, packing K pair {kt*32+2v+16h, +1}.
// A fragment read is then 8 contiguous dwords = 2x global_load_b128.
__global__ __launch_bounds__(256) void pack_b(const float* __restrict__ B,
                                              const float* __restrict__ Bm, int ldb,
                                              int K, int N,
                                              u32* __restrict__ Bh, u32* __restrict__ Bl) {
  size_t total = (size_t)(K / 32) * N * 16;
  size_t g = (size_t)blockIdx.x * 256 + threadIdx.x;
  if (g >= total) return;
  int v    = (int)(g & 7);
  int half = (int)((g >> 3) & 1);
  size_t rest = g >> 4;             // = kt*N + n
  int n  = (int)(rest % N);
  int kt = (int)(rest / N);
  int k  = kt * 32 + 2 * v + 16 * half;
  size_t o0 = (size_t)k * ldb + n;
  size_t o1 = o0 + ldb;
  float f0 = B[o0] * Bm[o0];
  float f1 = B[o1] * Bm[o1];
  H2 hi, lo;
  split1(f0, hi.f[0], lo.f[0]);
  split1(f1, hi.f[1], lo.f[1]);
  Bh[g] = hi.u;
  Bl[g] = lo.u;
}

// ---- fragment loaders from packed planes ----------------------------------
// A planes: row-major f16 [M][K]; uint4 row stride = K/8.
__device__ __forceinline__ void load_afrag(const uint4* __restrict__ P,
                                           int row, int K, int kt, int half, Frag& f) {
  size_t base = (size_t)row * (K >> 3) + (size_t)kt * 4 + 2 * half;
  f.u4[0] = P[base];
  f.u4[1] = P[base + 1];
}
// B planes: swizzled; uint4 index = ((kt*N + n)*2 + half)*2.
__device__ __forceinline__ void load_bfrag(const uint4* __restrict__ P,
                                           int kt, int N, int n, int half, Frag& f) {
  size_t base = (((size_t)kt * N + n) * 2 + half) * 2;
  f.u4[0] = P[base];
  f.u4[1] = P[base + 1];
}

// ===========================================================================
// GEMM: C[M,N] = sum over <=2 passes of Ap @ Bp + bias.  No LDS: packed
// fragments stream from L2.  WG tile 256x128, 8 waves (4x2), wave 64x64
// (4x4 fragments -> 48 WMMAs : 32 b128 loads per K-tile).
// ===========================================================================
#define TM 256
#define TN 128

__global__ __launch_bounds__(256) void sdnn_gemm_packed(
    const uint4* __restrict__ Ah0, const uint4* __restrict__ Al0, int K0,
    const uint4* __restrict__ Bh0, const uint4* __restrict__ Bl0,
    const uint4* __restrict__ Ah1, const uint4* __restrict__ Al1, int K1,
    const uint4* __restrict__ Bh1, const uint4* __restrict__ Bl1,
    const float* __restrict__ bias,
    float* __restrict__ Cg, int ldc, int N)
{
  const int tid  = threadIdx.x;
  const int m0   = blockIdx.y * TM;
  const int n0   = blockIdx.x * TN;
  const int w    = tid >> 5;
  const int lane = tid & 31;
  const int half = lane >> 4;
  const int lm   = lane & 15;
  const int wm   = (w >> 1) * 64;   // wave row offset in WG tile
  const int wn   = (w & 1) * 64;    // wave col offset in WG tile

  v8f acc[4][4];
#pragma unroll
  for (int i = 0; i < 4; ++i)
#pragma unroll
    for (int j = 0; j < 4; ++j)
      acc[i][j] = (v8f){0.f,0.f,0.f,0.f,0.f,0.f,0.f,0.f};

  // clamped column indices for fragment loads (stores are guarded separately)
  int ncol[4];
#pragma unroll
  for (int j = 0; j < 4; ++j) {
    int n = n0 + wn + 16 * j + lm;
    ncol[j] = n < N ? n : N - 1;
  }

  for (int pass = 0; pass < 2; ++pass) {
    const uint4* Ah = pass ? Ah1 : Ah0;
    const uint4* Al = pass ? Al1 : Al0;
    const uint4* Bh = pass ? Bh1 : Bh0;
    const uint4* Bl = pass ? Bl1 : Bl0;
    const int K     = pass ? K1  : K0;
    if (!Ah) continue;

    const int nkt = K >> 5;
    for (int kt = 0; kt < nkt; ++kt) {
      Frag ah[4], al[4], bh[4], bl[4];
#pragma unroll
      for (int i = 0; i < 4; ++i) {
        const int row = m0 + wm + 16 * i + lm;
        load_afrag(Ah, row, K, kt, half, ah[i]);
        load_afrag(Al, row, K, kt, half, al[i]);
        if (kt + 1 < nkt)
          __builtin_prefetch(Ah + (size_t)row * (K >> 3) + (size_t)(kt + 1) * 4, 0, 1);
      }
#pragma unroll
      for (int j = 0; j < 4; ++j) {
        load_bfrag(Bh, kt, N, ncol[j], half, bh[j]);
        load_bfrag(Bl, kt, N, ncol[j], half, bl[j]);
      }
#pragma unroll
      for (int i = 0; i < 4; ++i)
#pragma unroll
        for (int j = 0; j < 4; ++j)
          acc[i][j] = wmma3(ah[i].h, al[i].h, bh[j].h, bl[j].h, acc[i][j]);
    }
  }

  // epilogue: C/D layout (M = v + 8*half, N = lane%16)
#pragma unroll
  for (int i = 0; i < 4; ++i)
#pragma unroll
    for (int j = 0; j < 4; ++j) {
      const int n = n0 + wn + 16 * j + lm;
      if (n < N) {
        const float bz = bias ? bias[n] : 0.f;
#pragma unroll
        for (int v = 0; v < 8; ++v) {
          const int m = m0 + wm + 16 * i + v + 8 * half;
          Cg[(size_t)m * ldc + n] = acc[i][j][v] + bz;
        }
      }
    }
}

// ===========================================================================
// Recurrence: per workgroup 64 exclusive batch rows, 64 column tiles of 16.
// History S = hidden[:, :16t] @ w2m[:16t, tile] via packed split-f16 WMMA
// (K over-read to 32 safe: future hidden cols pre-zeroed in the planes);
// then 16 sequential scalar steps for the in-tile triangular part, which
// emit the hidden hi/lo f16 planes directly.
// ===========================================================================
__global__ __launch_bounds__(128) void sdnn_recurrence(
    const float* __restrict__ xw1,   // [8192][1024] fp32
    const float* __restrict__ w2,    // [2048][2048] fp32 (for exact in-tile block)
    const float* __restrict__ m2,
    const uint4* __restrict__ W2h,   // packed w2m planes, K=1024, N=1024
    const uint4* __restrict__ W2l,
    _Float16* __restrict__ hh,       // hidden hi plane [8192][1024]
    _Float16* __restrict__ hl)       // hidden lo plane [8192][1024]
{
  __shared__ float Ss[64 * 17];
  __shared__ float w2in[16 * 17];

  const int tid  = threadIdx.x;
  const int row0 = blockIdx.x * 64;
  const int w    = tid >> 5;
  const int lane = tid & 31;
  const int half = lane >> 4;
  const int lm   = lane & 15;

  // pre-zero our 64 rows of both planes (over-read safety)
  {
    uint4 z = {0u, 0u, 0u, 0u};
    uint4* zh = reinterpret_cast<uint4*>(hh + (size_t)row0 * 1024);
    uint4* zl = reinterpret_cast<uint4*>(hl + (size_t)row0 * 1024);
    for (int i = tid; i < 64 * 1024 / 8; i += 128) { zh[i] = z; zl[i] = z; }
  }
  __syncthreads();

  const uint4* HH = reinterpret_cast<const uint4*>(hh);
  const uint4* HL = reinterpret_cast<const uint4*>(hl);
  const int arow = row0 + 16 * w + lm;

  for (int t = 0; t < 64; ++t) {
    const int j0 = t * 16;
    v8f acc = (v8f){0.f,0.f,0.f,0.f,0.f,0.f,0.f,0.f};

    const int nchunks = (t + 1) >> 1;  // ceil(16t/32)
    for (int kt = 0; kt < nchunks; ++kt) {
      Frag ah, al, bh, bl;
      load_afrag(HH, arow, 1024, kt, half, ah);
      load_afrag(HL, arow, 1024, kt, half, al);
      load_bfrag(W2h, kt, 1024, j0 + lm, half, bh);
      load_bfrag(W2l, kt, 1024, j0 + lm, half, bl);
      acc = wmma3(ah.h, al.h, bh.h, bl.h, acc);
    }

    // stage S fragment (M = v + 8*half in wave row block) and in-tile w2
#pragma unroll
    for (int v = 0; v < 8; ++v)
      Ss[(16 * w + v + 8 * half) * 17 + lm] = acc[v];
    for (int e = tid; e < 256; e += 128) {
      const int mm = e >> 4, nn = e & 15;
      const size_t o = (size_t)(j0 + mm) * 2048 + j0 + nn;
      w2in[mm * 17 + nn] = w2[o] * m2[o];
    }
    __syncthreads();

    // sequential triangular tail: thread r owns batch row r
    if (tid < 64) {
      const int r = tid;
      const size_t xo = (size_t)(row0 + r) * 1024 + j0;
      float h[16];
#pragma unroll
      for (int l = 0; l < 16; ++l) {
        float v = xw1[xo + l] + Ss[r * 17 + l];
        for (int mI = 0; mI < l; ++mI) v += h[mI] * w2in[mI * 17 + l];
        h[l] = v > 0.f ? v : 0.f;
      }
#pragma unroll
      for (int l = 0; l < 16; ++l) {
        _Float16 hv = (_Float16)h[l];
        hh[xo + l] = hv;
        hl[xo + l] = (_Float16)(h[l] - (float)hv);
      }
    }
    __syncthreads();  // new hidden cols visible to next tile's WMMA
  }
}

// ===========================================================================
// Launch
// ===========================================================================
extern "C" void kernel_launch(void* const* d_in, const int* in_sizes, int n_in,
                              void* d_out, int out_size, void* d_ws, size_t ws_size,
                              hipStream_t stream) {
  const float* x  = (const float*)d_in[0];
  const float* w1 = (const float*)d_in[1];
  const float* w2 = (const float*)d_in[2];
  const float* w3 = (const float*)d_in[3];
  const float* w4 = (const float*)d_in[4];
  const float* b1 = (const float*)d_in[5];
  const float* b2 = (const float*)d_in[6];
  const float* m1 = (const float*)d_in[7];
  const float* m2 = (const float*)d_in[8];
  const float* m3 = (const float*)d_in[9];
  const float* m4 = (const float*)d_in[10];
  float* out = (float*)d_out;

  // ---- workspace carve-up (~152 MB total) ----
  char* ws = (char*)d_ws;
  float* xw1 = (float*)ws;            ws += (size_t)8192 * 1024 * 4;   // 32 MB
  u32* xh  = (u32*)ws;                ws += (size_t)8192 * 2048 * 2;   // 32 MB
  u32* xl  = (u32*)ws;                ws += (size_t)8192 * 2048 * 2;   // 32 MB
  u32* hhp = (u32*)ws;                ws += (size_t)8192 * 1024 * 2;   // 16 MB
  u32* hlp = (u32*)ws;                ws += (size_t)8192 * 1024 * 2;   // 16 MB
  u32* w1h = (u32*)ws;                ws += (size_t)2048 * 1024 * 2;   //  4 MB
  u32* w1l = (u32*)ws;                ws += (size_t)2048 * 1024 * 2;
  u32* w2h = (u32*)ws;                ws += (size_t)1024 * 1024 * 2;   //  2 MB
  u32* w2l = (u32*)ws;                ws += (size_t)1024 * 1024 * 2;
  u32* w3h = (u32*)ws;                ws += (size_t)1024 * 1000 * 2;   //  2 MB
  u32* w3l = (u32*)ws;                ws += (size_t)1024 * 1000 * 2;
  u32* w4h = (u32*)ws;                ws += (size_t)2048 * 1000 * 2;   //  4 MB
  u32* w4l = (u32*)ws;                ws += (size_t)2048 * 1000 * 2;

  // ---- pack operands once ----
  {
    size_t n4 = (size_t)8192 * 2048 / 4;
    pack_a<<<dim3((unsigned)((n4 + 255) / 256)), 256, 0, stream>>>(x, xh, xl, n4);
  }
  auto launch_pack_b = [&](const float* B, const float* Bm, int ldb, int K, int N,
                           u32* Bh, u32* Bl) {
    size_t total = (size_t)(K / 32) * N * 16;
    pack_b<<<dim3((unsigned)((total + 255) / 256)), 256, 0, stream>>>(B, Bm, ldb, K, N, Bh, Bl);
  };
  launch_pack_b(w1, m1, 2048, 2048, 1024, w1h, w1l);  // w1[:, :1024]
  launch_pack_b(w2, m2, 2048, 1024, 1024, w2h, w2l);  // w2[:1024, :1024]
  launch_pack_b(w3, m3, 1000, 1024, 1000, w3h, w3l);  // w3[:1024, :]
  launch_pack_b(w4, m4, 1000, 2048, 1000, w4h, w4l);

  // (1) xw1 = x @ (w1*m1)[:, :1024] + b1[:1024]
  sdnn_gemm_packed<<<dim3(1024 / TN, 8192 / TM), 256, 0, stream>>>(
      (const uint4*)xh, (const uint4*)xl, 2048, (const uint4*)w1h, (const uint4*)w1l,
      nullptr, nullptr, 0, nullptr, nullptr,
      b1, xw1, 1024, 1024);

  // (2) column recurrence over 1024 columns (emits hidden hi/lo planes)
  sdnn_recurrence<<<dim3(8192 / 64), 128, 0, stream>>>(
      xw1, w2, m2, (const uint4*)w2h, (const uint4*)w2l,
      (_Float16*)hhp, (_Float16*)hlp);

  // (3) out = hidden @ (w3*m3) + x @ (w4*m4) + b2
  sdnn_gemm_packed<<<dim3((1000 + TN - 1) / TN, 8192 / TM), 256, 0, stream>>>(
      (const uint4*)hhp, (const uint4*)hlp, 1024, (const uint4*)w3h, (const uint4*)w3l,
      (const uint4*)xh, (const uint4*)xl, 2048, (const uint4*)w4h, (const uint4*)w4l,
      b2, out, 1000, 1000);
}
